// SpatialClassifierVN_66434554134836
// MI455X (gfx1250) — compile-verified, matrix operand order checked by reading, not run on hardware
//
#include <hip/hip_runtime.h>
#include <hip/hip_bf16.h>

// ---------------------------------------------------------------------------
// SpatialClassifierVN forward for gfx1250 (MI455X, wave32, WMMA).
// Dense per-row GEMMs on v_wmma_f32_16x16x32_bf16 (f32 accumulate), 16-row
// tiles per 128-thread workgroup. Explicit address spaces (AS3 LDS / AS1
// global); fragments loaded as contiguous b128 runs (one wait per fragment);
// all GEMM shapes are compile-time template parameters (no integer division,
// fully unrolled K loops).
// ---------------------------------------------------------------------------

typedef __attribute__((ext_vector_type(16))) __bf16 v16bf;
typedef __attribute__((ext_vector_type(8)))  float  v8f;
typedef __attribute__((ext_vector_type(4)))  float  f4;

typedef __attribute__((address_space(3))) float*        LDSP;
typedef const __attribute__((address_space(3))) float*  CLDSP;
typedef const __attribute__((address_space(1))) float*  GLBP;
typedef __attribute__((address_space(1))) float*        GLBPW;
typedef const __attribute__((address_space(1))) int*    GLBI;
typedef const __attribute__((address_space(3))) f4*     CLDS4;
typedef const __attribute__((address_space(1))) f4*     GLB4;

__device__ __forceinline__ LDSP  to_lds(float* p)        { return (LDSP)(unsigned)(unsigned long long)p; }
__device__ __forceinline__ GLBP  to_glb(const float* p)  { return (GLBP)(unsigned long long)p; }
__device__ __forceinline__ GLBPW to_glbw(float* p)       { return (GLBPW)(unsigned long long)p; }
__device__ __forceinline__ GLBI  to_glbi(const int* p)   { return (GLBI)(unsigned long long)p; }

#define NQc    4096
#define NCc    10000
#define EKNNc  (NQc * 32)
#define EQc    65536
#define EAc    262144
#define NHc    4
#define NCLSc  13
#define PIc    3.14159265358979323846f

constexpr int BLK = 128;   // 4 waves of 32

// ---------------------------------------------------------------------------
// wave-level 16x16 WMMA tile: C(16x16) += A(16xK) * W(16 rows, K cols)^T
// Both A (LDS) and W (global) are row-major f32 with row stride K.
// Per 32-K step each lane needs two contiguous 8-float runs -> 4x b128 loads
// per operand, one wait, then packed bf16 converts.
// ---------------------------------------------------------------------------
template<int K>
__device__ __forceinline__ v8f wmma_tile(CLDSP A, GLBP W, v8f acc)
{
  const int lane = threadIdx.x & 31;
  const int r = lane & 15, h = lane >> 4;
#pragma unroll
  for (int k0 = 0; k0 < K; k0 += 32) {
    CLDS4 a1 = (CLDS4)(A + r * K + k0 + 8 * h);
    CLDS4 a2 = (CLDS4)(A + r * K + k0 + 16 + 8 * h);
    GLB4  b1 = (GLB4)(W + r * K + k0 + 8 * h);
    GLB4  b2 = (GLB4)(W + r * K + k0 + 16 + 8 * h);
    f4 av0 = a1[0], av1 = a1[1], av2 = a2[0], av3 = a2[1];
    f4 bv0 = b1[0], bv1 = b1[1], bv2 = b2[0], bv3 = b2[1];
    v16bf a, b;
#pragma unroll
    for (int t = 0; t < 4; ++t) {
      a[t]      = (__bf16)av0[t];  a[4 + t]  = (__bf16)av1[t];
      a[8 + t]  = (__bf16)av2[t];  a[12 + t] = (__bf16)av3[t];
      b[t]      = (__bf16)bv0[t];  b[4 + t]  = (__bf16)bv1[t];
      b[8 + t]  = (__bf16)bv2[t];  b[12 + t] = (__bf16)bv3[t];
    }
    acc = __builtin_amdgcn_wmma_f32_16x16x32_bf16(false, a, false, b,
                                                  (short)0, acc, false, false);
  }
  return acc;
}

// block-level GEMM: C[16][N] = A[16][K] * W[N][K]^T (+bias). N % 16 == 0.
template<int N, int K>
__device__ __forceinline__ void block_gemm(CLDSP A, GLBP W, GLBP bias, LDSP C)
{
  const int wid = threadIdx.x >> 5, nw = BLK >> 5;
  for (int nt = wid; nt < N / 16; nt += nw) {
    v8f acc = {};
    acc = wmma_tile<K>(A, W + nt * 16 * K, acc);
    const int lane = threadIdx.x & 31;
    const int n = nt * 16 + (lane & 15), h = lane >> 4;
    const float bv = bias ? bias[n] : 0.f;
#pragma unroll
    for (int r = 0; r < 8; ++r) C[(r + 8 * h) * N + n] = acc[r] + bv;
  }
}

// scalar fallback for tiny N (13 / 4 / 1 outputs)
template<int N, int K>
__device__ __forceinline__ void block_gemm_small(CLDSP A, GLBP W, LDSP C)
{
  for (int idx = threadIdx.x; idx < 16 * N; idx += BLK) {
    int r = idx / N, n = idx % N;
    float s = 0.f;
#pragma unroll 4
    for (int k = 0; k < K; ++k) s += A[r * K + k] * W[n * K + k];
    C[r * N + n] = s;
  }
}

struct GVLP { const float *wv1, *wv2, *ws, *gw, *gb; };

// ---------------------------------------------------------------------------
// GVLinear over a 16-row tile, everything in LDS; shapes compile-time.
//   vec_in : [3][16][IVEC]   coord-major
//   cat    : [16][HID+ISCA]  cols [HID, HID+ISCA) prefilled with scalars
//   vinter : [3][16][HID]    scratch
//   out_s  : [16][OSCA]
//   out_v  : [3][16][OVEC]   (gated)
// Caller must __syncthreads() after filling vec_in / cat scalars.
// ---------------------------------------------------------------------------
template<int IVEC, int HID, int ISCA, int OSCA, int OVEC>
__device__ void gvlinear_b(GVLP P, CLDSP vec_in, LDSP cat,
                           LDSP vinter, LDSP out_s, LDSP out_v)
{
  constexpr int KCAT = HID + ISCA;
  const int tid = threadIdx.x, wid = tid >> 5, nw = BLK >> 5;
  GLBP wv1 = to_glb(P.wv1), wv2 = to_glb(P.wv2), ws = to_glb(P.ws);
  GLBP gw = to_glb(P.gw), gb = to_glb(P.gb);
  // 1) vinter[c] = vec_in[c] @ wv1^T    (N=HID, K=IVEC)  -- WMMA
  {
    constexpr int T = HID / 16;
    for (int j = wid; j < 3 * T; j += nw) {
      int c = j / T, nt = j % T;
      v8f acc = {};
      acc = wmma_tile<IVEC>(vec_in + c * 16 * IVEC, wv1 + nt * 16 * IVEC, acc);
      const int lane = tid & 31;
      const int n = nt * 16 + (lane & 15), h = lane >> 4;
#pragma unroll
      for (int r = 0; r < 8; ++r)
        vinter[(c * 16 + r + 8 * h) * HID + n] = acc[r];
    }
  }
  __syncthreads();
  // 2) vnorm -> cat[:, 0:HID]
  for (int idx = tid; idx < 16 * HID; idx += BLK) {
    int r = idx / HID, hc = idx % HID;
    float x = vinter[(0 * 16 + r) * HID + hc];
    float y = vinter[(1 * 16 + r) * HID + hc];
    float z = vinter[(2 * 16 + r) * HID + hc];
    cat[r * KCAT + hc] = sqrtf(x * x + y * y + z * z);
  }
  __syncthreads();
  // 3) out_s = cat @ ws^T   (N=OSCA, K=HID+ISCA)
  if constexpr ((OSCA & 15) == 0) block_gemm<OSCA, KCAT>(cat, ws, (GLBP)0, out_s);
  else                            block_gemm_small<OSCA, KCAT>(cat, ws, out_s);
  // 4) out_v[c] = vinter[c] @ wv2^T  (N=OVEC, K=HID)
  if constexpr ((OVEC & 15) == 0) {
    constexpr int T = OVEC / 16;
    for (int j = wid; j < 3 * T; j += nw) {
      int c = j / T, nt = j % T;
      v8f acc = {};
      acc = wmma_tile<HID>(vinter + c * 16 * HID, wv2 + nt * 16 * HID, acc);
      const int lane = tid & 31;
      const int n = nt * 16 + (lane & 15), h = lane >> 4;
#pragma unroll
      for (int r = 0; r < 8; ++r)
        out_v[(c * 16 + r + 8 * h) * OVEC + n] = acc[r];
    }
  } else {
    for (int idx = tid; idx < 3 * 16 * OVEC; idx += BLK) {
      int c = idx / (16 * OVEC), rem = idx % (16 * OVEC);
      int r = rem / OVEC, o = rem % OVEC;
      float s = 0.f;
#pragma unroll 4
      for (int k = 0; k < HID; ++k)
        s += vinter[(c * 16 + r) * HID + k] * wv2[o * HID + k];
      out_v[(c * 16 + r) * OVEC + o] = s;
    }
  }
  __syncthreads();
  // 5) gate = sigmoid(out_s @ gw^T + gb); out_v *= gate
  for (int idx = tid; idx < 16 * OVEC; idx += BLK) {
    int r = idx / OVEC, o = idx % OVEC;
    float g = gb[o];
#pragma unroll 4
    for (int k = 0; k < OSCA; ++k) g += out_s[r * OSCA + k] * gw[o * OSCA + k];
    g = 1.f / (1.f + expf(-g));
    out_v[(0 * 16 + r) * OVEC + o] *= g;
    out_v[(1 * 16 + r) * OVEC + o] *= g;
    out_v[(2 * 16 + r) * OVEC + o] *= g;
  }
  __syncthreads();
}

__device__ __forceinline__ void leaky_s(LDSP s, int n)
{
  for (int i = threadIdx.x; i < n; i += BLK) {
    float x = s[i]; s[i] = x >= 0.f ? x : 0.01f * x;
  }
  __syncthreads();
}

// VN leaky relu: d = v @ wdir^T (WMMA), then per-channel projection update
template<int OVEC>
__device__ void gvp_act(const float* wdir_g, LDSP v, LDSP d)
{
  const int tid = threadIdx.x, wid = tid >> 5, nw = BLK >> 5;
  GLBP wdir = to_glb(wdir_g);
  {
    constexpr int T = OVEC / 16;
    for (int j = wid; j < 3 * T; j += nw) {
      int c = j / T, nt = j % T;
      v8f acc = {};
      acc = wmma_tile<OVEC>(v + c * 16 * OVEC, wdir + nt * 16 * OVEC, acc);
      const int lane = tid & 31;
      const int n = nt * 16 + (lane & 15), h = lane >> 4;
#pragma unroll
      for (int r = 0; r < 8; ++r)
        d[(c * 16 + r + 8 * h) * OVEC + n] = acc[r];
    }
  }
  __syncthreads();
  for (int idx = tid; idx < 16 * OVEC; idx += BLK) {
    int r = idx / OVEC, o = idx % OVEC;
    float x0 = v[(0 * 16 + r) * OVEC + o], x1 = v[(16 + r) * OVEC + o], x2 = v[(32 + r) * OVEC + o];
    float d0 = d[(0 * 16 + r) * OVEC + o], d1 = d[(16 + r) * OVEC + o], d2 = d[(32 + r) * OVEC + o];
    float dot = x0 * d0 + x1 * d1 + x2 * d2;
    if (dot < 0.f) {
      float dn = d0 * d0 + d1 * d1 + d2 * d2;
      float f = 0.8f * dot / (dn + 1e-6f);
      x0 -= f * d0; x1 -= f * d1; x2 -= f * d2;
    }
    v[(0 * 16 + r) * OVEC + o] = x0; v[(16 + r) * OVEC + o] = x1; v[(32 + r) * OVEC + o] = x2;
  }
  __syncthreads();
}

// order-preserving float<->uint for atomicMax
__device__ __forceinline__ unsigned fenc(float f) {
  unsigned u = __float_as_uint(f);
  return (u & 0x80000000u) ? ~u : (u | 0x80000000u);
}
__device__ __forceinline__ float fdec(unsigned u) {
  u = (u & 0x80000000u) ? (u & 0x7FFFFFFFu) : ~u;
  return __uint_as_float(u);
}

// ============================ Kernel A: node GVL ============================
struct NodeArgs {
  const float *sca, *vec; GVLP p; float *ns, *nv;
};
constexpr int A_SM = (3*16*64 + 16*320 + 3*16*64 + 16*128 + 3*16*32);

__global__ __launch_bounds__(BLK) void k_node_gvl(NodeArgs a)
{
  extern __shared__ float smraw[];
  LDSP sm = to_lds(smraw);
  LDSP vec_in = sm;                      // 3*16*64
  LDSP cat    = vec_in + 3*16*64;        // 16*320
  LDSP vinter = cat + 16*320;            // 3*16*64
  LDSP out_s  = vinter + 3*16*64;        // 16*128
  LDSP out_v  = out_s + 16*128;          // 3*16*32
  GLBP gvec = to_glb(a.vec), gsca = to_glb(a.sca);
  GLBPW gns = to_glbw(a.ns), gnv = to_glbw(a.nv);
  const int row0 = blockIdx.x * 16;
  for (int idx = threadIdx.x; idx < 16*64*3; idx += BLK) {
    int r = idx / 192, rem = idx % 192, ch = rem / 3, c = rem % 3;
    vec_in[(c*16 + r) * 64 + ch] = gvec[((size_t)(row0+r)*64 + ch)*3 + c];
  }
  for (int idx = threadIdx.x; idx < 16*256; idx += BLK) {
    int r = idx / 256, ch = idx % 256;
    cat[r*320 + 64 + ch] = gsca[(size_t)(row0+r)*256 + ch];
  }
  __syncthreads();
  gvlinear_b<64, 64, 256, 128, 32>(a.p, vec_in, cat, vinter, out_s, out_v);
  for (int idx = threadIdx.x; idx < 16*128; idx += BLK) {
    int r = idx / 128, ch = idx % 128;
    gns[(size_t)(row0+r)*128 + ch] = out_s[r*128 + ch];
  }
  for (int idx = threadIdx.x; idx < 16*96; idx += BLK) {
    int r = idx / 96, k = idx % 96, o = k / 3, c = k % 3;
    gnv[(size_t)(row0+r)*96 + o*3 + c] = out_v[(c*16 + r)*32 + o];
  }
}

// ============================ Kernel B: knn messages ========================
struct MsgArgs {
  const float *pos_q, *pos_c; const int* eknn; const float* vexp;
  GVLP pedge; const float* edge_vnd;
  const float *ns, *nv;
  const float *sca_w, *sca_b, *e2n_w, *e2n_b, *n2e_w, *n2e_b, *evn;
  GVLP pout;
  float *y_s, *y_v;
};
constexpr int B_ev_in = 0;
constexpr int B_cat1  = B_ev_in + 3072;
constexpr int B_vin1  = B_cat1 + 2048;
constexpr int B_es    = B_vin1 + 3072;
constexpr int B_ev    = B_es + 1024;
constexpr int B_d1    = B_ev + 3072;
constexpr int B_ns    = B_d1 + 3072;
constexpr int B_nv    = B_ns + 2048;
constexpr int B_ys    = B_nv + 1536;
constexpr int B_t1    = B_ys + 2048;
constexpr int B_t2    = B_t1 + 512;
constexpr int B_ev2   = B_t2 + 512;
constexpr int B_yv    = B_ev2 + 1536;
constexpr int B_cat2  = B_yv + 1536;
constexpr int B_vin2  = B_cat2 + 2560;
constexpr int B_os    = B_vin2 + 1536;
constexpr int B_ov    = B_os + 2048;
constexpr int B_misc  = B_ov + 1536;   // dist16, ux16, uy16, uz16, Cc16
constexpr int B_SM    = B_misc + 96;

__global__ __launch_bounds__(BLK) void k_msg(MsgArgs a)
{
  extern __shared__ float smraw[];
  LDSP sm = to_lds(smraw);
  __shared__ int scol[16], srow[16];
  LDSP ev_in = sm + B_ev_in; LDSP cat1 = sm + B_cat1; LDSP vin1 = sm + B_vin1;
  LDSP es = sm + B_es; LDSP ev = sm + B_ev; LDSP d1 = sm + B_d1;
  LDSP ns_l = sm + B_ns; LDSP nv_l = sm + B_nv; LDSP ys = sm + B_ys;
  LDSP t1 = sm + B_t1; LDSP t2 = sm + B_t2; LDSP ev2 = sm + B_ev2;
  LDSP yv = sm + B_yv; LDSP cat2 = sm + B_cat2; LDSP vin2 = sm + B_vin2;
  LDSP os = sm + B_os; LDSP ov = sm + B_ov;
  LDSP dist = sm + B_misc; LDSP ux = dist + 16; LDSP uy = ux + 16;
  LDSP uz = uy + 16; LDSP Cc = uz + 16;

  GLBI geknn = to_glbi(a.eknn);
  GLBP gposq = to_glb(a.pos_q), gposc = to_glb(a.pos_c), gvexp = to_glb(a.vexp);
  GLBP gns = to_glb(a.ns), gnv = to_glb(a.nv);

  const int e0 = blockIdx.x * 16;
  if (threadIdx.x < 16) {
    int e = e0 + threadIdx.x;
    int row = geknn[e], col = geknn[EKNNc + e];
    srow[threadIdx.x] = row; scol[threadIdx.x] = col;
    float vx = gposq[row*3+0] - gposc[col*3+0];
    float vy = gposq[row*3+1] - gposc[col*3+1];
    float vz = gposq[row*3+2] - gposc[col*3+2];
    float d = sqrtf(vx*vx + vy*vy + vz*vz);
    float inv = 1.f / (d + 1e-7f);
    dist[threadIdx.x] = d;
    ux[threadIdx.x] = vx*inv; uy[threadIdx.x] = vy*inv; uz[threadIdx.x] = vz*inv;
    Cc[threadIdx.x] = (d <= 10.f) ? 0.5f * (cosf(d * (PIc/10.f)) + 1.f) : 0.f;
  }
  __syncthreads();
  const float st = 10.f/63.f, cf = -0.5f/(st*st);
  for (int idx = threadIdx.x; idx < 16*64; idx += BLK) {
    int r = idx >> 6, ch = idx & 63;
    float dd = dist[r] - st * ch;
    cat1[r*128 + 64 + ch] = expf(cf * dd * dd);
  }
  for (int idx = threadIdx.x; idx < 3*16*64; idx += BLK) {
    int c = idx / 1024, rem = idx % 1024, r = rem >> 6, ch = rem & 63;
    float u = (c == 0) ? ux[r] : (c == 1 ? uy[r] : uz[r]);
    ev_in[(c*16 + r)*64 + ch] = u * gvexp[ch];
  }
  for (int idx = threadIdx.x; idx < 16*128; idx += BLK) {
    int r = idx / 128, ch = idx % 128;
    ns_l[r*128 + ch] = gns[(size_t)scol[r]*128 + ch];
  }
  for (int idx = threadIdx.x; idx < 16*96; idx += BLK) {
    int r = idx / 96, k = idx % 96, o = k / 3, c = k % 3;
    nv_l[(c*16 + r)*32 + o] = gnv[(size_t)scol[r]*96 + o*3 + c];
  }
  __syncthreads();
  gvlinear_b<64, 64, 64, 64, 64>(a.pedge, ev_in, cat1, vin1, es, ev);
  leaky_s(es, 16*64);
  gvp_act<64>(a.edge_vnd, ev, d1);
  block_gemm<128, 64>(es,   to_glb(a.sca_w), to_glb(a.sca_b), ys);
  block_gemm<32, 64>(es,    to_glb(a.e2n_w), to_glb(a.e2n_b), t1);
  block_gemm<32, 128>(ns_l, to_glb(a.n2e_w), to_glb(a.n2e_b), t2);
  for (int c = 0; c < 3; ++c)
    block_gemm<32, 64>(ev + c*16*64, to_glb(a.evn), (GLBP)0, ev2 + c*16*32);
  __syncthreads();
  for (int idx = threadIdx.x; idx < 16*128; idx += BLK) {
    int r = idx / 128, ch = idx % 128;
    cat2[r*160 + 32 + ch] = ys[r*128 + ch] * ns_l[r*128 + ch];
  }
  for (int idx = threadIdx.x; idx < 3*16*32; idx += BLK) {
    int c = idx / 512, rem = idx % 512, r = rem >> 5, o = rem & 31;
    yv[(c*16 + r)*32 + o] = t1[r*32 + o] * nv_l[(c*16 + r)*32 + o]
                          + t2[r*32 + o] * ev2[(c*16 + r)*32 + o];
  }
  __syncthreads();
  gvlinear_b<32, 32, 128, 128, 32>(a.pout, yv, cat2, vin2, os, ov);
  const int q = srow[0];   // 16 consecutive knn edges share one query row
  for (int ch = threadIdx.x; ch < 128; ch += BLK) {
    float s = 0.f;
#pragma unroll
    for (int r = 0; r < 16; ++r) s += os[r*128 + ch] * Cc[r];
    atomicAdd(&a.y_s[(size_t)q*128 + ch], s);
  }
  for (int t = threadIdx.x; t < 96; t += BLK) {
    int o = t / 3, c = t % 3;
    float s = 0.f;
#pragma unroll
    for (int r = 0; r < 16; ++r) s += ov[(c*16 + r)*32 + o] * Cc[r];
    atomicAdd(&a.y_v[(size_t)q*96 + t], s);
  }
}

// ============================ Kernel C: cls head ============================
struct ClsArgs { const float *ys, *yv; GVLP pgvp; const float* vnd; GVLP pgvl; float* out; };
constexpr int C_SM = 1536 + 2560 + 1536 + 2048 + 1536 + 1536 + 2560 + 1536 + 256 + 48;

__global__ __launch_bounds__(BLK) void k_cls(ClsArgs a)
{
  extern __shared__ float smraw[];
  LDSP sm = to_lds(smraw);
  LDSP vec1 = sm;              LDSP cat1 = vec1 + 1536;
  LDSP vin1 = cat1 + 2560;     LDSP s1 = vin1 + 1536;
  LDSP v1 = s1 + 2048;         LDSP d1 = v1 + 1536;
  LDSP cat2 = d1 + 1536;       LDSP vin2 = cat2 + 2560;
  LDSP s2 = vin2 + 1536;       LDSP v2 = s2 + 256;
  GLBP gys = to_glb(a.ys), gyv = to_glb(a.yv);
  GLBPW gout = to_glbw(a.out);
  const int row0 = blockIdx.x * 16;
  for (int idx = threadIdx.x; idx < 16*128; idx += BLK) {
    int r = idx / 128, ch = idx % 128;
    cat1[r*160 + 32 + ch] = gys[(size_t)(row0+r)*128 + ch];
  }
  for (int idx = threadIdx.x; idx < 16*96; idx += BLK) {
    int r = idx / 96, k = idx % 96, o = k / 3, c = k % 3;
    vec1[(c*16 + r)*32 + o] = gyv[(size_t)(row0+r)*96 + k];
  }
  __syncthreads();
  gvlinear_b<32, 32, 128, 128, 32>(a.pgvp, vec1, cat1, vin1, s1, v1);
  leaky_s(s1, 16*128);
  gvp_act<32>(a.vnd, v1, d1);
  for (int idx = threadIdx.x; idx < 16*128; idx += BLK) {
    int r = idx / 128, ch = idx % 128;
    cat2[r*160 + 32 + ch] = s1[r*128 + ch];
  }
  __syncthreads();
  gvlinear_b<32, 32, 128, NCLSc, 1>(a.pgvl, v1, cat2, vin2, s2, v2);
  for (int idx = threadIdx.x; idx < 16*NCLSc; idx += BLK) {
    int r = idx / NCLSc, ch = idx % NCLSc;
    gout[(size_t)(row0+r)*NCLSc + ch] = s2[r*NCLSc + ch];
  }
}

// ================== Kernel D: nn_edge + edge_feat (fused) ===================
struct EdgeFeatArgs {
  const float *pos_q, *pos_c; const int* eq; const float* vexp;
  GVLP pg1; const float* vnd1; GVLP pl1;
  const float *ys, *yv, *nsca, *nvec;
  GVLP pg2; const float* vnd2; GVLP pl2;
  float *eas, *eav;
};
constexpr int D_P  = 35328;
constexpr int D_SM = D_P + 2048 + 1536 + 64;

__global__ __launch_bounds__(BLK) void k_edgefeat(EdgeFeatArgs a)
{
  extern __shared__ float smraw[];
  LDSP sm = to_lds(smraw);
  __shared__ int si[16], sj[16];
  LDSP P   = sm;
  LDSP efs = P + D_P;          // 16*128
  LDSP efv = efs + 2048;       // 3*16*32
  LDSP bd  = efv + 1536; LDSP ux = bd + 16; LDSP uy = ux + 16; LDSP uz = uy + 16;
  GLBI geq = to_glbi(a.eq);
  GLBP gposq = to_glb(a.pos_q), gposc = to_glb(a.pos_c), gvexp = to_glb(a.vexp);
  GLBP gys = to_glb(a.ys), gyv = to_glb(a.yv);
  GLBP gnsca = to_glb(a.nsca), gnvec = to_glb(a.nvec);
  GLBPW geas = to_glbw(a.eas), geav = to_glbw(a.eav);
  const int e0 = blockIdx.x * 16;
  if (threadIdx.x < 16) {
    int e = e0 + threadIdx.x;
    int i = geq[e], j = geq[EQc + e];
    si[threadIdx.x] = i; sj[threadIdx.x] = j;
    float vx = gposq[i*3+0] - gposc[j*3+0];
    float vy = gposq[i*3+1] - gposc[j*3+1];
    float vz = gposq[i*3+2] - gposc[j*3+2];
    float d = sqrtf(vx*vx + vy*vy + vz*vz);
    float inv = 1.f / (d + 1e-7f);
    bd[threadIdx.x] = d; ux[threadIdx.x] = vx*inv; uy[threadIdx.x] = vy*inv; uz[threadIdx.x] = vz*inv;
  }
  __syncthreads();
  // ---- phase 1: nn_edge
  LDSP vec1 = P, cat1 = P+3072, vin1 = P+5120, s1 = P+8192,
       v1 = P+10240, d1 = P+11776, cat2 = P+13312, vin2 = P+15872;
  const float st3 = 3.f/63.f, cf3 = -0.5f/(st3*st3);
  for (int idx = threadIdx.x; idx < 16*64; idx += BLK) {
    int r = idx >> 6, ch = idx & 63;
    float dd = bd[r] - st3 * ch;
    cat1[r*128 + 64 + ch] = expf(cf3 * dd * dd);
  }
  for (int idx = threadIdx.x; idx < 3*16*64; idx += BLK) {
    int c = idx / 1024, rem = idx % 1024, r = rem >> 6, ch = rem & 63;
    float u = (c == 0) ? ux[r] : (c == 1 ? uy[r] : uz[r]);
    vec1[(c*16 + r)*64 + ch] = u * gvexp[ch];
  }
  __syncthreads();
  gvlinear_b<64, 64, 64, 128, 32>(a.pg1, vec1, cat1, vin1, s1, v1);
  leaky_s(s1, 16*128);
  gvp_act<32>(a.vnd1, v1, d1);
  for (int idx = threadIdx.x; idx < 16*128; idx += BLK) {
    int r = idx / 128, ch = idx % 128;
    cat2[r*160 + 32 + ch] = s1[r*128 + ch];
  }
  __syncthreads();
  gvlinear_b<32, 32, 128, 128, 32>(a.pl1, v1, cat2, vin2, efs, efv);
  // ---- phase 2: edge_feat (reuses arena P)
  LDSP cat3 = P, vec3 = P+10240, vin3 = P+16384, s3 = P+22528,
       v3 = P+24576, d3 = P+26112, cat4 = P+27648, vin4 = P+30208,
       eas = P+31744, eav = P+33792;
  for (int idx = threadIdx.x; idx < 16*128; idx += BLK) {
    int r = idx / 128, ch = idx % 128;
    cat3[r*640 + 128 + ch] = gys[(size_t)si[r]*128 + ch];
  }
  for (int idx = threadIdx.x; idx < 16*256; idx += BLK) {
    int r = idx / 256, ch = idx % 256;
    cat3[r*640 + 256 + ch] = gnsca[(size_t)sj[r]*256 + ch];
  }
  for (int idx = threadIdx.x; idx < 16*128; idx += BLK) {
    int r = idx / 128, ch = idx % 128;
    cat3[r*640 + 512 + ch] = efs[r*128 + ch];
  }
  for (int idx = threadIdx.x; idx < 16*96; idx += BLK) {
    int r = idx / 96, k = idx % 96, o = k / 3, c = k % 3;
    vec3[(c*16 + r)*128 + o] = gyv[(size_t)si[r]*96 + k];
  }
  for (int idx = threadIdx.x; idx < 16*192; idx += BLK) {
    int r = idx / 192, k = idx % 192, o = k / 3, c = k % 3;
    vec3[(c*16 + r)*128 + 32 + o] = gnvec[(size_t)sj[r]*192 + k];
  }
  for (int idx = threadIdx.x; idx < 3*16*32; idx += BLK) {
    int c = idx / 512, rem = idx % 512, r = rem >> 5, o = rem & 31;
    vec3[(c*16 + r)*128 + 96 + o] = efv[(c*16 + r)*32 + o];
  }
  __syncthreads();
  gvlinear_b<128, 128, 512, 128, 32>(a.pg2, vec3, cat3, vin3, s3, v3);
  leaky_s(s3, 16*128);
  gvp_act<32>(a.vnd2, v3, d3);
  for (int idx = threadIdx.x; idx < 16*128; idx += BLK) {
    int r = idx / 128, ch = idx % 128;
    cat4[r*160 + 32 + ch] = s3[r*128 + ch];
  }
  __syncthreads();
  gvlinear_b<32, 32, 128, 128, 32>(a.pl2, v3, cat4, vin4, eas, eav);
  for (int idx = threadIdx.x; idx < 16*128; idx += BLK) {
    int r = idx / 128, ch = idx % 128;
    geas[(size_t)(e0+r)*128 + ch] = eas[r*128 + ch];
  }
  for (int idx = threadIdx.x; idx < 16*96; idx += BLK) {
    int r = idx / 96, k = idx % 96, o = k / 3, c = k % 3;
    geav[(size_t)(e0+r)*96 + k] = eav[(c*16 + r)*32 + o];
  }
}

// ============================ Kernel E: q/k/v ===============================
struct QKVArgs {
  const float *eas, *eav; GVLP pq, pk, pv;
  float *qs, *qv, *ks, *kv, *vs, *vv;
};
constexpr int E_SM = 1536 + 2560 + 1536 + 2048 + 1536;

__global__ __launch_bounds__(BLK) void k_qkv(QKVArgs a)
{
  extern __shared__ float smraw[];
  LDSP sm = to_lds(smraw);
  LDSP vec1 = sm;            LDSP cat1 = vec1 + 1536;
  LDSP vin1 = cat1 + 2560;   LDSP outs = vin1 + 1536;
  LDSP outv = outs + 2048;
  GLBP geas = to_glb(a.eas), geav = to_glb(a.eav);
  const int n0 = blockIdx.x * 16;
  for (int idx = threadIdx.x; idx < 16*128; idx += BLK) {
    int r = idx / 128, ch = idx % 128;
    cat1[r*160 + 32 + ch] = geas[(size_t)(n0+r)*128 + ch];
  }
  for (int idx = threadIdx.x; idx < 16*96; idx += BLK) {
    int r = idx / 96, k = idx % 96, o = k / 3, c = k % 3;
    vec1[(c*16 + r)*32 + o] = geav[(size_t)(n0+r)*96 + k];
  }
  __syncthreads();
  GVLP ps[3] = { a.pq, a.pk, a.pv };
  float* gs[3] = { a.qs, a.ks, a.vs };
  float* gv[3] = { a.qv, a.kv, a.vv };
  for (int m = 0; m < 3; ++m) {
    gvlinear_b<32, 32, 128, 128, 32>(ps[m], vec1, cat1, vin1, outs, outv);
    GLBPW os = to_glbw(gs[m]), ov = to_glbw(gv[m]);
    for (int idx = threadIdx.x; idx < 16*128; idx += BLK) {
      int r = idx / 128, ch = idx % 128;
      os[(size_t)(n0+r)*128 + ch] = outs[r*128 + ch];
    }
    for (int idx = threadIdx.x; idx < 16*96; idx += BLK) {
      int r = idx / 96, k = idx % 96, o = k / 3, c = k % 3;
      ov[(size_t)(n0+r)*96 + k] = outv[(c*16 + r)*32 + o];
    }
    __syncthreads();
  }
}

// ======================= Kernel F: attention scores =========================
struct AttScoreArgs {
  const float* pos_c; const int *iatt, *tri; const float* trif;
  const float* bvw; GVLP pbias;
  const float *qs, *qv, *ks, *kv;
  float *ss, *sv; unsigned *ms, *mv;
};
constexpr int F_SM = 1536 + 2560 + 1536 + 64 + 192 + 64;

__global__ __launch_bounds__(BLK) void k_attsc(AttScoreArgs a)
{
  extern __shared__ float smraw[];
  LDSP sm = to_lds(smraw);
  __shared__ int sii[16], sjj[16];
  LDSP vec1 = sm;            LDSP cat1 = vec1 + 1536;
  LDSP vin1 = cat1 + 2560;   LDSP bs = vin1 + 1536;
  LDSP bv = bs + 64;
  LDSP bd = bv + 192; LDSP ux = bd + 16; LDSP uy = ux + 16; LDSP uz = uy + 16;
  GLBI giatt = to_glbi(a.iatt), gtri = to_glbi(a.tri);
  GLBP gposc = to_glb(a.pos_c), gtrif = to_glb(a.trif), gbvw = to_glb(a.bvw);
  GLBP gqs = to_glb(a.qs), gqv = to_glb(a.qv), gks = to_glb(a.ks), gkv = to_glb(a.kv);
  GLBPW gss = to_glbw(a.ss), gsv = to_glbw(a.sv);
  const int a0 = blockIdx.x * 16;
  if (threadIdx.x < 16) {
    int e = a0 + threadIdx.x;
    sii[threadIdx.x] = giatt[e]; sjj[threadIdx.x] = giatt[EAc + e];
    int na = gtri[e], nb = gtri[EAc + e];
    float bx = gposc[na*3+0] - gposc[nb*3+0];
    float by = gposc[na*3+1] - gposc[nb*3+1];
    float bz = gposc[na*3+2] - gposc[nb*3+2];
    float d = sqrtf(bx*bx + by*by + bz*bz);
    float inv = 1.f / (d + 1e-7f);
    bd[threadIdx.x] = d; ux[threadIdx.x] = bx*inv; uy[threadIdx.x] = by*inv; uz[threadIdx.x] = bz*inv;
  }
  __syncthreads();
  const float stg = 10.f/122.f, cfg = -0.5f/(stg*stg);
  for (int idx = threadIdx.x; idx < 16*128; idx += BLK) {
    int r = idx / 128, ch = idx % 128;
    float v;
    if (ch < 123) { float dd = bd[r] - stg * ch; v = expf(cfg * dd * dd); }
    else          { v = gtrif[(size_t)(a0+r)*5 + (ch - 123)]; }
    cat1[r*160 + 32 + ch] = v;
  }
  for (int idx = threadIdx.x; idx < 3*16*32; idx += BLK) {
    int c = idx / 512, rem = idx % 512, r = rem >> 5, o = rem & 31;
    float u = (c == 0) ? ux[r] : (c == 1 ? uy[r] : uz[r]);
    vec1[(c*16 + r)*32 + o] = u * gbvw[o];
  }
  __syncthreads();
  gvlinear_b<32, 32, 128, NHc, NHc>(a.pbias, vec1, cat1, vin1, bs, bv);
  for (int t = threadIdx.x; t < 64; t += BLK) {
    int r = t >> 2, h = t & 3;
    int e = a0 + r, ii = sii[r], jj = sjj[r];
    float qks = 0.f;
#pragma unroll
    for (int c = 0; c < 32; ++c)
      qks += gqs[(size_t)ii*128 + h*32 + c] * gks[(size_t)jj*128 + h*32 + c];
    float qkv = 0.f;
#pragma unroll
    for (int k = 0; k < 24; ++k)
      qkv += gqv[(size_t)ii*96 + h*24 + k] * gkv[(size_t)jj*96 + h*24 + k];
    float b0 = bv[(0*16 + r)*4 + h], b1 = bv[(16 + r)*4 + h], b2 = bv[(32 + r)*4 + h];
    float s_s = bs[r*4 + h] + qks;
    float s_v = (b0*b0 + b1*b1 + b2*b2) + qkv;
    gss[(size_t)e*4 + h] = s_s; gsv[(size_t)e*4 + h] = s_v;
    atomicMax(&a.ms[ii*4 + h], fenc(s_s));
    atomicMax(&a.mv[ii*4 + h], fenc(s_v));
  }
}

// ====================== Kernel G: exp + segment sum =========================
__global__ __launch_bounds__(256) void k_attexp(const int* iatt, float* ss, float* sv,
                                               const unsigned* ms, const unsigned* mv,
                                               float* sums, float* sumv)
{
  int idx = blockIdx.x * 256 + threadIdx.x;
  if (idx >= EAc * NHc) return;
  int e = idx >> 2, h = idx & 3;
  int ii = to_glbi(iatt)[e];
  float es = expf(to_glb(ss)[idx] - fdec(ms[ii*4 + h]));
  float ev = expf(to_glb(sv)[idx] - fdec(mv[ii*4 + h]));
  to_glbw(ss)[idx] = es; to_glbw(sv)[idx] = ev;
  atomicAdd(&sums[ii*4 + h], es);
  atomicAdd(&sumv[ii*4 + h], ev);
}

// ================== Kernel H: weighted value scatter ========================
__global__ __launch_bounds__(256) void k_attacc(const int* iatt,
                                               const float* ss, const float* sv,
                                               const float* sums, const float* sumv,
                                               const float* vs, const float* vv,
                                               float* accs, float* accv)
{
  long idx = (long)blockIdx.x * 256 + threadIdx.x;
  if (idx >= (long)EAc * 224) return;
  int e = (int)(idx / 224), t = (int)(idx % 224);
  GLBI gi = to_glbi(iatt);
  int ii = gi[e], jj = gi[EAc + e];
  if (t < 128) {
    int h = t >> 5;
    float alpha = to_glb(ss)[(size_t)e*4 + h] / (to_glb(sums)[ii*4 + h] + 1e-16f);
    atomicAdd(&accs[(size_t)ii*128 + t], alpha * to_glb(vs)[(size_t)jj*128 + t]);
  } else {
    int tv = t - 128, h = tv / 24;
    float alpha = to_glb(sv)[(size_t)e*4 + h] / (to_glb(sumv)[ii*4 + h] + 1e-16f);
    atomicAdd(&accv[(size_t)ii*96 + tv], alpha * to_glb(vv)[(size_t)jj*96 + tv]);
  }
}

// ================== Kernel I: residual + LN + edge_pred =====================
struct OutArgs {
  const float *eas, *eav, *accs, *accv;
  const float *gs, *bs, *gv, *bv;
  GVLP pred; float* out;
};
constexpr int I_SM = 1536 + 2560 + 1536 + 64 + 48 + 2048 + 1536;

__global__ __launch_bounds__(BLK) void k_final(OutArgs a)
{
  extern __shared__ float smraw[];
  LDSP sm = to_lds(smraw);
  LDSP vec1 = sm;            LDSP cat1 = vec1 + 1536;
  LDSP vin1 = cat1 + 2560;   LDSP s2 = vin1 + 1536;
  LDSP v2 = s2 + 64;         LDSP lsr = v2 + 48;
  LDSP lvr = lsr + 2048;
  GLBP geas = to_glb(a.eas), geav = to_glb(a.eav);
  GLBP gaccs = to_glb(a.accs), gaccv = to_glb(a.accv);
  GLBP ggs = to_glb(a.gs), gbs = to_glb(a.bs), ggv = to_glb(a.gv), gbv = to_glb(a.bv);
  GLBPW gout = to_glbw(a.out);
  const int row0 = blockIdx.x * 16;
  if (threadIdx.x < 16) {
    int r = threadIdx.x, n = row0 + r;
    // scalar LN over 128
    float mean = 0.f;
    for (int ch = 0; ch < 128; ++ch) {
      float x = geas[(size_t)n*128 + ch] + gaccs[(size_t)n*128 + ch];
      lsr[r*128 + ch] = x; mean += x;
    }
    mean *= (1.f/128.f);
    float var = 0.f;
    for (int ch = 0; ch < 128; ++ch) { float dv = lsr[r*128 + ch] - mean; var += dv*dv; }
    var *= (1.f/128.f);
    float inv = rsqrtf(var + 1e-5f);
    for (int ch = 0; ch < 128; ++ch)
      cat1[r*160 + 32 + ch] = (lsr[r*128 + ch] - mean) * inv * ggs[ch] + gbs[ch];
    // vector LN over (32,3)
    float vm = 0.f;
    for (int k = 0; k < 96; ++k) {
      float x = geav[(size_t)n*96 + k] + gaccv[(size_t)n*96 + k];
      lvr[r*96 + k] = x; vm += x;
    }
    vm *= (1.f/96.f);
    float vvar = 0.f;
    for (int k = 0; k < 96; ++k) { float dv = lvr[r*96 + k] - vm; vvar += dv*dv; }
    vvar *= (1.f/96.f);
    float vinv = rsqrtf(vvar + 1e-5f);
    for (int k = 0; k < 96; ++k) {
      int o = k / 3, c = k % 3;
      vec1[(c*16 + r)*32 + o] = (lvr[r*96 + k] - vm) * vinv * ggv[k] + gbv[k];
    }
  }
  __syncthreads();
  gvlinear_b<32, 32, 128, 4, 1>(a.pred, vec1, cat1, vin1, s2, v2);
  for (int idx = threadIdx.x; idx < 16*4; idx += BLK) {
    int r = idx >> 2, t = idx & 3;
    gout[(size_t)(row0+r)*4 + t] = s2[r*4 + t];
  }
}

// ============================================================================
// Host launcher. Param pointers follow jax pytree (sorted-key) flatten order.
// ============================================================================
extern "C" void kernel_launch(void* const* d_in, const int* in_sizes, int n_in,
                              void* d_out, int out_size, void* d_ws, size_t ws_size,
                              hipStream_t stream)
{
  const float* pos_q = (const float*)d_in[0];
  const float* pos_c = (const float*)d_in[1];
  const float* nsca  = (const float*)d_in[2];
  const float* nvec  = (const float*)d_in[3];
  const int*   eknn  = (const int*)d_in[4];
  const int*   eq    = (const int*)d_in[5];
  const int*   iatt  = (const int*)d_in[6];
  const int*   tri   = (const int*)d_in[7];
  const float* trif  = (const float*)d_in[8];
  const float* const* P = (const float* const*)(d_in + 9);

  auto mk = [&](int i) {
    GVLP p; p.gb = P[i]; p.gw = P[i+1]; p.wv1 = P[i+2]; p.wv2 = P[i+3]; p.ws = P[i+4];
    return p;
  };
  // attn
  GVLP p_bias = mk(0);                    // attn.bias_gvl
  const float* bias_vec_w = P[5];
  GVLP p_k = mk(6);
  const float *ln_s_b = P[11], *ln_s_g = P[12], *ln_v_b = P[13], *ln_v_g = P[14];
  GVLP p_q = mk(15), p_v = mk(20);
  // cls
  GVLP p_cls_gvl = mk(25), p_cls_gvp = mk(30);
  const float* cls_vnd = P[35];
  // edge_feat
  GVLP p_ef_gvl = mk(36), p_ef_gvp = mk(41);
  const float* ef_vnd = P[46];
  // edge_pred
  GVLP p_pred = mk(47);
  // msg
  const float *e2n_b = P[52], *e2n_w = P[53];
  GVLP p_edge = mk(54); const float* edge_vnd = P[59];
  const float* evn = P[60];
  const float *n2e_b = P[61], *n2e_w = P[62];
  GVLP p_node = mk(63), p_out = mk(68);
  const float *sca_b = P[73], *sca_w = P[74];
  // nn_edge
  GVLP p_nn_gvl = mk(75), p_nn_gvp = mk(80);
  const float* nn_vnd = P[85];
  const float* vexp = P[86];

  // ---- workspace carve ----
  float* w = (float*)d_ws;
  auto alloc = [&](size_t n) { float* p = w; w += n; return p; };
  float* ns   = alloc((size_t)NCc * 128);
  float* nv   = alloc((size_t)NCc * 96);
  float* ys   = alloc((size_t)NQc * 128);
  float* yv   = alloc((size_t)NQc * 96);
  float* eas  = alloc((size_t)EQc * 128);
  float* eav  = alloc((size_t)EQc * 96);
  float* qs   = alloc((size_t)EQc * 128);
  float* qv   = alloc((size_t)EQc * 96);
  float* ks   = alloc((size_t)EQc * 128);
  float* kv   = alloc((size_t)EQc * 96);
  float* vs   = alloc((size_t)EQc * 128);
  float* vv   = alloc((size_t)EQc * 96);
  float* ss   = alloc((size_t)EAc * 4);
  float* sv   = alloc((size_t)EAc * 4);
  unsigned* ms = (unsigned*)alloc((size_t)EQc * 4);
  unsigned* mv = (unsigned*)alloc((size_t)EQc * 4);
  float* sums = alloc((size_t)EQc * 4);
  float* sumv = alloc((size_t)EQc * 4);
  float* accs = alloc((size_t)EQc * 128);
  float* accv = alloc((size_t)EQc * 96);

  hipMemsetAsync(ys,   0, (size_t)NQc * 128 * 4, stream);
  hipMemsetAsync(yv,   0, (size_t)NQc * 96 * 4, stream);
  hipMemsetAsync(ms,   0, (size_t)EQc * 4 * 4, stream);
  hipMemsetAsync(mv,   0, (size_t)EQc * 4 * 4, stream);
  hipMemsetAsync(sums, 0, (size_t)EQc * 4 * 4, stream);
  hipMemsetAsync(sumv, 0, (size_t)EQc * 4 * 4, stream);
  hipMemsetAsync(accs, 0, (size_t)EQc * 128 * 4, stream);
  hipMemsetAsync(accv, 0, (size_t)EQc * 96 * 4, stream);

  float* ycls  = (float*)d_out;
  float* epred = ycls + (size_t)NQc * NCLSc;

  { NodeArgs a{nsca, nvec, p_node, ns, nv};
    k_node_gvl<<<NCc/16, BLK, A_SM*4, stream>>>(a); }

  { MsgArgs a{pos_q, pos_c, eknn, vexp, p_edge, edge_vnd, ns, nv,
              sca_w, sca_b, e2n_w, e2n_b, n2e_w, n2e_b, evn, p_out, ys, yv};
    k_msg<<<EKNNc/16, BLK, B_SM*4, stream>>>(a); }

  { ClsArgs a{ys, yv, p_cls_gvp, cls_vnd, p_cls_gvl, ycls};
    k_cls<<<NQc/16, BLK, C_SM*4, stream>>>(a); }

  { EdgeFeatArgs a{pos_q, pos_c, eq, vexp, p_nn_gvp, nn_vnd, p_nn_gvl,
                   ys, yv, nsca, nvec, p_ef_gvp, ef_vnd, p_ef_gvl, eas, eav};
    k_edgefeat<<<EQc/16, BLK, D_SM*4, stream>>>(a); }

  { QKVArgs a{eas, eav, p_q, p_k, p_v, qs, qv, ks, kv, vs, vv};
    k_qkv<<<EQc/16, BLK, E_SM*4, stream>>>(a); }

  { AttScoreArgs a{pos_c, iatt, tri, trif, bias_vec_w, p_bias,
                   qs, qv, ks, kv, ss, sv, ms, mv};
    k_attsc<<<EAc/16, BLK, F_SM*4, stream>>>(a); }

  k_attexp<<<(EAc*NHc + 255)/256, 256, 0, stream>>>(iatt, ss, sv, ms, mv, sums, sumv);

  k_attacc<<<(unsigned)(((long)EAc*224 + 255)/256), 256, 0, stream>>>(
      iatt, ss, sv, sums, sumv, vs, vv, accs, accv);

  { OutArgs a{eas, eav, accs, accv, ln_s_g, ln_s_b, ln_v_g, ln_v_b, p_pred, epred};
    k_final<<<EQc/16, BLK, I_SM*4, stream>>>(a); }

  (void)in_sizes; (void)n_in; (void)out_size; (void)ws_size;
}
